// TemporalLobe_63977832841684
// MI455X (gfx1250) — matrix operand order repaired and checked by reading, not compiled, not run on hardware
//
#include <hip/hip_runtime.h>
#include <hip/hip_bf16.h>

#define B_  512
#define H_  1024
#define V_  32000
#define T_  20

typedef __attribute__((ext_vector_type(16))) __bf16 v16bf;
typedef __attribute__((ext_vector_type(8)))  float  v8f;

union Frag16 { v16bf v; uint4 q[2]; };

__device__ __forceinline__ unsigned short f2bf(float f) {
    unsigned u = __float_as_uint(f);
    u += 0x7FFFu + ((u >> 16) & 1u);          // round-to-nearest-even
    return (unsigned short)(u >> 16);
}

// monotonic key: larger float -> larger unsigned
__device__ __forceinline__ unsigned fkey(float f) {
    unsigned u = __float_as_uint(f);
    return (u & 0x80000000u) ? ~u : (u | 0x80000000u);
}

__device__ __forceinline__ unsigned long long shfl_xor_u64(unsigned long long x, int m) {
    unsigned lo = __shfl_xor((unsigned)x, m, 32);
    unsigned hi = __shfl_xor((unsigned)(x >> 32), m, 32);
    return ((unsigned long long)hi << 32) | lo;
}

// ---- software-pipelined wave tile: 16(M) x 64(N), K stepped by 32, double-buffered ----
// A layout (16-bit A 16x32):  half 0 -> K{0..7,16..23}, half 1 -> K{8..15,24..31}
// B layout (16-bit B 32x16):  half 0 -> K{0..7,16..23}, half 1 -> K{8..15,24..31} (same as A)
__device__ __forceinline__ void wave_gemm_16x64(
        const unsigned short* __restrict__ A, const unsigned short* __restrict__ W,
        int K, int mRow, int nCol0, int half, v8f acc[4]) {
    const unsigned short* aP  = A + (size_t)mRow  * K + half * 8;
    const unsigned short* wP0 = W + (size_t)nCol0 * K + half * 8;

    Frag16 a0, a1, b0[4], b1[4];
    auto loadA = [&](Frag16& f, int k0) {
        f.q[0] = *reinterpret_cast<const uint4*>(aP + k0);
        f.q[1] = *reinterpret_cast<const uint4*>(aP + k0 + 16);
    };
    auto loadB = [&](Frag16* f, int k0) {
#pragma unroll
        for (int t = 0; t < 4; ++t) {
            const unsigned short* w = wP0 + (size_t)t * 16 * K + k0;
            f[t].q[0] = *reinterpret_cast<const uint4*>(w);
            f[t].q[1] = *reinterpret_cast<const uint4*>(w + 16);
        }
    };
    auto mm = [&](Frag16& a, Frag16* b) {
#pragma unroll
        for (int t = 0; t < 4; ++t)
            acc[t] = __builtin_amdgcn_wmma_f32_16x16x32_bf16(
                         false, a.v, false, b[t].v, (short)0, acc[t], false, false);
    };

    loadA(a0, 0); loadB(b0, 0);                 // prologue
    for (int k0 = 0; k0 < K - 64; k0 += 64) {
        loadA(a1, k0 + 32); loadB(b1, k0 + 32); // prefetch stage 1
        mm(a0, b0);                             // compute stage 0 (overlaps with loads)
        loadA(a0, k0 + 64); loadB(b0, k0 + 64); // prefetch stage 0'
        mm(a1, b1);                             // compute stage 1
    }
    loadA(a1, K - 32); loadB(b1, K - 32);       // epilogue
    mm(a0, b0);
    mm(a1, b1);
}

// ---------------- one-time conversions / init ----------------

__global__ void cvt_f32_bf16(const float* __restrict__ src,
                             unsigned short* __restrict__ dst, int n) {
    int idx = (blockIdx.x * blockDim.x + threadIdx.x) * 4;
    if (idx >= n) return;
    float4 f = *reinterpret_cast<const float4*>(src + idx);
    ushort4 o;
    o.x = f2bf(f.x); o.y = f2bf(f.y); o.z = f2bf(f.z); o.w = f2bf(f.w);
    *reinterpret_cast<ushort4*>(dst + idx) = o;
}

__global__ void init_state(const float* __restrict__ latent,
                           float* __restrict__ h, unsigned short* __restrict__ hb,
                           int* __restrict__ tok, unsigned long long* __restrict__ rowBest) {
    int i = blockIdx.x * blockDim.x + threadIdx.x;
    if (i < B_ * H_) { float f = latent[i]; h[i] = f; hb[i] = f2bf(f); }
    if (i < B_)      { tok[i] = 0; rowBest[i] = 0ull; }   // SOS = 0
}

// ---------------- per-step kernels ----------------

__global__ void gather_emb(const unsigned short* __restrict__ table,
                           const int* __restrict__ tok,
                           unsigned short* __restrict__ emb) {
    int i = blockIdx.x * blockDim.x + threadIdx.x;   // over B_*H_/8
    int b = i >> 7;                                  // H_/8 = 128
    int c = i & 127;
    const uint4* srow = reinterpret_cast<const uint4*>(table + (size_t)tok[b] * H_);
    reinterpret_cast<uint4*>(emb + (size_t)b * H_)[c] = srow[c];
}

// C[M,N] = A[M,K] * W[N,K]^T + bias ; bf16 in, f32 out.
// block = 8 waves arranged 4(M) x 2(N) -> block tile 64 x 128 (max W reuse).
__global__ __launch_bounds__(256) void gemm_bf16_nt(
        const unsigned short* __restrict__ A, const unsigned short* __restrict__ W,
        const float* __restrict__ bias, float* __restrict__ C,
        int N, int K) {
    const int wave = threadIdx.x >> 5, lane = threadIdx.x & 31;
    const int half = lane >> 4, l15 = lane & 15;
    const int mBase = blockIdx.y * 64  + (wave >> 1) * 16;
    const int nBase = blockIdx.x * 128 + (wave & 1) * 64;

    v8f acc[4] = {};
    wave_gemm_16x64(A, W, K, mBase + l15, nBase + l15, half, acc);

#pragma unroll
    for (int t = 0; t < 4; ++t) {
        int n = nBase + t * 16 + l15;
        float bv = bias[n];
#pragma unroll
        for (int r = 0; r < 8; ++r) {
            int m = mBase + r + half * 8;   // C layout: vgpr r, half0 M=r, half1 M=r+8
            C[(size_t)m * N + n] = acc[t][r] + bv;
        }
    }
}

// PyTorch GRU cell gate fusion: r,z,n -> h_new ; writes f32 + bf16 hidden
__global__ void gru_fuse(const float* __restrict__ gx, const float* __restrict__ gh,
                         float* __restrict__ h, unsigned short* __restrict__ hb) {
    int i = blockIdx.x * blockDim.x + threadIdx.x;
    if (i >= B_ * H_) return;
    int b = i >> 10, j = i & 1023;
    size_t base = (size_t)b * 3 * H_ + j;
    float xr = gx[base], xz = gx[base + H_], xn = gx[base + 2 * H_];
    float hr = gh[base], hz = gh[base + H_], hn = gh[base + 2 * H_];
    float r = 1.f / (1.f + __expf(-(xr + hr)));
    float z = 1.f / (1.f + __expf(-(xz + hz)));
    float n = tanhf(xn + r * hn);
    float hnew = (1.f - z) * n + z * h[i];
    h[i]  = hnew;
    hb[i] = f2bf(hnew);
}

// logits = h @ w_out^T + b_out with fused per-row argmax (64 MB/step of logits never stored)
__global__ __launch_bounds__(256) void logits_argmax(
        const unsigned short* __restrict__ Hb, const unsigned short* __restrict__ Wo,
        const float* __restrict__ bout, unsigned long long* __restrict__ rowBest) {
    const int wave = threadIdx.x >> 5, lane = threadIdx.x & 31;
    const int half = lane >> 4, l15 = lane & 15;
    const int mBase = blockIdx.y * 64  + (wave >> 1) * 16;
    const int nBase = blockIdx.x * 128 + (wave & 1) * 64;

    v8f acc[4] = {};
    wave_gemm_16x64(Hb, Wo, H_, mBase + l15, nBase + l15, half, acc);

    // per-row argmax: for fixed (r, half) all 4 accumulators + 16 lanes hold one C row
#pragma unroll
    for (int r = 0; r < 8; ++r) {
        unsigned long long best = 0ull;
#pragma unroll
        for (int t = 0; t < 4; ++t) {
            int n = nBase + t * 16 + l15;
            float v = acc[t][r] + bout[n];
            unsigned long long key =
                ((unsigned long long)fkey(v) << 32) | (unsigned)(0xFFFFFFFFu - (unsigned)n);
            best = key > best ? key : best;
        }
#pragma unroll
        for (int m = 1; m <= 8; m <<= 1) {     // reduction stays inside the 16-lane half
            unsigned long long o = shfl_xor_u64(best, m);
            best = o > best ? o : best;
        }
        if (l15 == 0)
            atomicMax(rowBest + (mBase + r + half * 8), best);
    }
}

__global__ void select_tok(unsigned long long* __restrict__ rowBest,
                           int* __restrict__ tok, int* __restrict__ out, int t) {
    int b = threadIdx.x;   // blockDim = 512
    unsigned long long k = rowBest[b];
    int v = (int)(0xFFFFFFFFu - (unsigned)(k & 0xFFFFFFFFu));
    tok[b] = v;
    out[(size_t)b * T_ + t] = v;   // output is [B, T]
    rowBest[b] = 0ull;             // rearm for next step
}

// ---------------- launch ----------------

extern "C" void kernel_launch(void* const* d_in, const int* in_sizes, int n_in,
                              void* d_out, int out_size, void* d_ws, size_t ws_size,
                              hipStream_t stream) {
    const float* latent    = (const float*)d_in[0];
    const float* embedding = (const float*)d_in[1];
    const float* w_ih      = (const float*)d_in[2];
    const float* w_hh      = (const float*)d_in[3];
    const float* b_ih      = (const float*)d_in[4];
    const float* b_hh      = (const float*)d_in[5];
    const float* w_out     = (const float*)d_in[6];
    const float* b_out     = (const float*)d_in[7];
    int* out = (int*)d_out;

    char* p = (char*)d_ws;
    unsigned short* embB = (unsigned short*)p;      p += (size_t)B_ * H_ * 2;
    float*          hF   = (float*)p;               p += (size_t)B_ * H_ * 4;
    unsigned short* hB   = (unsigned short*)p;      p += (size_t)B_ * H_ * 2;
    float*          gx   = (float*)p;               p += (size_t)B_ * 3 * H_ * 4;
    float*          gh   = (float*)p;               p += (size_t)B_ * 3 * H_ * 4;
    unsigned long long* rowBest = (unsigned long long*)p; p += (size_t)B_ * 8;
    int*            tok  = (int*)p;                 p += (size_t)B_ * 4;
    unsigned short* tabB = (unsigned short*)p;      p += (size_t)V_ * H_ * 2;
    unsigned short* wihB = (unsigned short*)p;      p += (size_t)3 * H_ * H_ * 2;
    unsigned short* whhB = (unsigned short*)p;      p += (size_t)3 * H_ * H_ * 2;
    unsigned short* woB  = (unsigned short*)p;      p += (size_t)V_ * H_ * 2;

    auto cvt = [&](const float* s, unsigned short* d, int n) {
        cvt_f32_bf16<<<(n / 4 + 255) / 256, 256, 0, stream>>>(s, d, n);
    };
    cvt(embedding, tabB, V_ * H_);
    cvt(w_ih, wihB, 3 * H_ * H_);
    cvt(w_hh, whhB, 3 * H_ * H_);
    cvt(w_out, woB, V_ * H_);
    init_state<<<(B_ * H_ + 255) / 256, 256, 0, stream>>>(latent, hF, hB, tok, rowBest);

    for (int t = 0; t < T_; ++t) {
        gather_emb<<<(B_ * H_ / 8) / 256, 256, 0, stream>>>(tabB, tok, embB);
        gemm_bf16_nt<<<dim3(3 * H_ / 128, B_ / 64), 256, 0, stream>>>(embB, wihB, b_ih, gx, 3 * H_, H_);
        gemm_bf16_nt<<<dim3(3 * H_ / 128, B_ / 64), 256, 0, stream>>>(hB,  whhB, b_hh, gh, 3 * H_, H_);
        gru_fuse<<<(B_ * H_ + 255) / 256, 256, 0, stream>>>(gx, gh, hF, hB);
        logits_argmax<<<dim3(V_ / 128, B_ / 64), 256, 0, stream>>>(hB, woB, b_out, rowBest);
        select_tok<<<1, B_, 0, stream>>>(rowBest, tok, out, t);
    }
}
